// FrameSelectorSMART_53360673685580
// MI455X (gfx1250) — matrix operand order repaired and checked by reading, not compiled, not run on hardware
//
#include <hip/hip_runtime.h>
#include <stdint.h>

#define T_FRAMES 16384
#define D_FEAT   2048
#define H_DIM    512
#define C_CLS    400
#define G_ROWS   2048   // 4*H
#define LSTM_BLOCKS 16

typedef __attribute__((ext_vector_type(16))) __bf16 v16bf;
typedef __attribute__((ext_vector_type(8)))  __bf16 v8bf;
typedef __attribute__((ext_vector_type(8)))  float  v8f;

static __device__ __forceinline__ float bf2f(unsigned short u) {
  union { unsigned int i; float f; } v; v.i = ((unsigned int)u) << 16; return v.f;
}
static __device__ __forceinline__ unsigned short f2bf(float f) {
  union { float f; unsigned int i; } v; v.f = f;
  unsigned int r = v.i + 0x7fffu + ((v.i >> 16) & 1u);   // round-to-nearest-even
  return (unsigned short)(r >> 16);
}
static __device__ __forceinline__ float sigm(float x) { return 1.f / (1.f + __expf(-x)); }

// ---------------- init / conversion / small kernels ----------------

__global__ void k_init(float* scalars, unsigned* ctrl, float* hcur) {
  int t = threadIdx.x;
  if (t < 16) scalars[t] = 0.f;
  if (t < 2)  ctrl[t] = 0u;
  if (t < H_DIM) hcur[t] = 0.f;
}

__global__ void k_f32_to_bf16(const float* __restrict__ src, unsigned short* __restrict__ dst, int n) {
  int i = blockIdx.x * blockDim.x + threadIdx.x;
  if (i < n) dst[i] = f2bf(src[i]);
}

__global__ void k_biassum(const float* __restrict__ a, const float* __restrict__ b,
                          float* __restrict__ out, int n) {
  int i = blockIdx.x * blockDim.x + threadIdx.x;
  if (i < n) out[i] = a[i] + b[i];
}

// sigmoid(dot(X[frame,:], Wv) + *addPtr + bias[0]); one wave per frame (wave32)
__global__ void __launch_bounds__(256) k_rowdot_sigmoid(
    const float* __restrict__ X, const float* __restrict__ Wv,
    const float* __restrict__ addPtr, const float* __restrict__ biasPtr,
    float* __restrict__ out, int D)
{
  int lane  = threadIdx.x & 31;
  int frame = blockIdx.x * 8 + (threadIdx.x >> 5);
  const float* xr = X + (size_t)frame * D;
  float acc = 0.f;
  for (int i = lane; i < D; i += 32) acc = fmaf(xr[i], Wv[i], acc);
#pragma unroll
  for (int m = 16; m > 0; m >>= 1) acc += __shfl_xor(acc, m, 32);
  if (lane == 0) {
    float add = (addPtr ? addPtr[0] : 0.f) + (biasPtr ? biasPtr[0] : 0.f);
    out[frame] = sigm(acc + add);
  }
}

__global__ void __launch_bounds__(256) k_reduce_sum(const float* __restrict__ v, int n, float* outp) {
  __shared__ float sh[256];
  float a = 0.f;
  for (int i = threadIdx.x; i < n; i += 256) a += v[i];
  sh[threadIdx.x] = a; __syncthreads();
  for (int s = 128; s > 0; s >>= 1) { if (threadIdx.x < s) sh[threadIdx.x] += sh[threadIdx.x + s]; __syncthreads(); }
  if (threadIdx.x == 0) outp[0] = sh[0];
}

// *outp = dot(a,b)/denom + bias[0]
__global__ void __launch_bounds__(256) k_dot_div_bias(
    const float* __restrict__ a, const float* __restrict__ b, int n,
    const float* denomPtr, float denomConst, const float* biasPtr, float* outp)
{
  __shared__ float sh[256];
  float s = 0.f;
  for (int i = threadIdx.x; i < n; i += 256) s = fmaf(a[i], b[i], s);
  sh[threadIdx.x] = s; __syncthreads();
  for (int t = 128; t > 0; t >>= 1) { if (threadIdx.x < t) sh[threadIdx.x] += sh[threadIdx.x + t]; __syncthreads(); }
  if (threadIdx.x == 0) {
    float den = denomPtr ? denomPtr[0] : denomConst;
    outp[0] = sh[0] / den + (biasPtr ? biasPtr[0] : 0.f);
  }
}

// zacc[d] = sum_t alpha[t]*x[t,d]  (deterministic, coalesced over d)
__global__ void __launch_bounds__(256) k_zacc(const float* __restrict__ x,
                                              const float* __restrict__ alpha,
                                              float* __restrict__ zacc)
{
  int d = blockIdx.x * blockDim.x + threadIdx.x;
  float acc = 0.f;
  for (int t = 0; t < T_FRAMES; ++t) acc = fmaf(alpha[t], x[(size_t)t * D_FEAT + d], acc);
  zacc[d] = acc;
}

// w = cumsum_t(beta*x); also bf16 copy and per-channel total (for z'')
__global__ void __launch_bounds__(256) k_wscan(const float* __restrict__ x, const float* __restrict__ beta,
                                               float* __restrict__ w, unsigned short* __restrict__ wbf,
                                               float* __restrict__ wsum)
{
  int d = blockIdx.x * blockDim.x + threadIdx.x;
  float acc = 0.f, ws = 0.f;
  for (int t = 0; t < T_FRAMES; ++t) {
    size_t idx = (size_t)t * D_FEAT + d;
    acc = fmaf(beta[t], x[idx], acc);
    w[idx] = acc; wbf[idx] = f2bf(acc); ws += acc;
  }
  wsum[d] = ws;
}

// cpre = cumsum_t(gamma*h) -> bf16 for classifier GEMM
__global__ void __launch_bounds__(256) k_cscan(const float* __restrict__ h, const float* __restrict__ gamma,
                                               unsigned short* __restrict__ cbf)
{
  int d = blockIdx.x * blockDim.x + threadIdx.x;   // 0..511
  float acc = 0.f;
  for (int t = 0; t < T_FRAMES; ++t) {
    size_t idx = (size_t)t * H_DIM + d;
    acc = fmaf(gamma[t], h[idx], acc);
    cbf[idx] = f2bf(acc);
  }
}

// ---------------- WMMA bf16 GEMM (TN: C[M,N] = A[M,K] * B[N,K]^T + bias[N]) ----------------
// A fragment (16x32 bf16): lanes 0-15 hold row m0+r, K = k..k+7 and k+16..k+23;
// lanes 16-31 hold K = k+8..k+15 and k+24..k+31 (ISA 7.12.2).
static __device__ __forceinline__ v16bf load_a_frag(const unsigned short* A, int lda, int m, int k, int lane) {
  int half = lane >> 4, r = lane & 15;
  const __bf16* p = reinterpret_cast<const __bf16*>(A) + (size_t)(m + r) * lda + k + half * 8;
  v8bf lo = *reinterpret_cast<const v8bf*>(p);
  v8bf hi = *reinterpret_cast<const v8bf*>(p + 16);
  return __builtin_shufflevector(lo, hi, 0,1,2,3,4,5,6,7,8,9,10,11,12,13,14,15);
}
// B fragment (32x16, K-major per lane): lane (half,c) holds column n0+c, K = k+half*8..+7 and k+16+half*8..+7
static __device__ __forceinline__ v16bf load_b_frag(const unsigned short* B, int ldb, int n, int k, int lane, int N) {
  int half = lane >> 4, c = lane & 15;
  int nr = n + c; if (nr > N - 1) nr = N - 1;            // clamp (edge tiles)
  const __bf16* p = reinterpret_cast<const __bf16*>(B) + (size_t)nr * ldb + k + half * 8;
  v8bf lo = *reinterpret_cast<const v8bf*>(p);
  v8bf hi = *reinterpret_cast<const v8bf*>(p + 16);
  return __builtin_shufflevector(lo, hi, 0,1,2,3,4,5,6,7,8,9,10,11,12,13,14,15);
}
// C/D layout: VGPR j -> row m0+j (lanes 0-15) / m0+8+j (lanes 16-31), col n0+(lane&15)
static __device__ __forceinline__ void store_c(float* C, int ldc, int m0, int n0, int lane,
                                               v8f acc, const float* bias, int N) {
  int half = lane >> 4, c = lane & 15;
  int n = n0 + c; if (n >= N) return;
  float b = bias ? bias[n] : 0.f;
#pragma unroll
  for (int j = 0; j < 8; ++j) {
    int m = m0 + half * 8 + j;
    C[(size_t)m * ldc + n] = acc[j] + b;
  }
}

__global__ void __launch_bounds__(256) k_gemm_bf16_tn(
    const unsigned short* __restrict__ A, const unsigned short* __restrict__ B,
    const float* __restrict__ bias, float* __restrict__ C, int M, int N, int K)
{
  int lane = threadIdx.x & 31;
  int wave = threadIdx.x >> 5;                    // 8 waves: 2 (M) x 4 (N)
  int m0 = blockIdx.y * 64  + (wave >> 2) * 32;
  int n0 = blockIdx.x * 128 + (wave & 3) * 32;
  v8f c00 = {}, c01 = {}, c10 = {}, c11 = {};
  for (int k = 0; k < K; k += 32) {
    v16bf a0 = load_a_frag(A, K, m0,      k, lane);
    v16bf a1 = load_a_frag(A, K, m0 + 16, k, lane);
    v16bf b0 = load_b_frag(B, K, n0,      k, lane, N);
    v16bf b1 = load_b_frag(B, K, n0 + 16, k, lane, N);
    c00 = __builtin_amdgcn_wmma_f32_16x16x32_bf16(false, a0, false, b0, (short)0, c00, false, false);
    c01 = __builtin_amdgcn_wmma_f32_16x16x32_bf16(false, a0, false, b1, (short)0, c01, false, false);
    c10 = __builtin_amdgcn_wmma_f32_16x16x32_bf16(false, a1, false, b0, (short)0, c10, false, false);
    c11 = __builtin_amdgcn_wmma_f32_16x16x32_bf16(false, a1, false, b1, (short)0, c11, false, false);
  }
  store_c(C, N, m0,      n0,      lane, c00, bias, N);
  store_c(C, N, m0,      n0 + 16, lane, c01, bias, N);
  store_c(C, N, m0 + 16, n0,      lane, c10, bias, N);
  store_c(C, N, m0 + 16, n0 + 16, lane, c11, bias, N);
}

// ---------------- multi-workgroup LSTM recurrence ----------------

static __device__ void global_barrier(unsigned* cnt, unsigned* gen, unsigned nblocks) {
  __syncthreads();
  if (threadIdx.x == 0) {
    __threadfence();
    unsigned g = __atomic_load_n(gen, __ATOMIC_RELAXED);
    if (atomicAdd(cnt, 1u) == nblocks - 1u) {
      __atomic_store_n(cnt, 0u, __ATOMIC_RELAXED);
      __threadfence();
      atomicAdd(gen, 1u);
    } else {
      while (__atomic_load_n(gen, __ATOMIC_RELAXED) == g) {
        asm volatile("s_sleep 1" ::: "memory");
      }
    }
    __threadfence();
  }
  __syncthreads();
}

// 16 blocks x 256 threads; block b owns hidden units [b*32, b*32+32) -> 128 gate rows.
// Two threads per gate row (split-K over 512), bf16 Whh resident in L2.
__global__ void __launch_bounds__(256) k_lstm(
    const float* __restrict__ gates, const unsigned short* __restrict__ Whh,
    float* __restrict__ hcur, float* __restrict__ hout,
    unsigned* cnt, unsigned* gen)
{
  __shared__ float hloc[H_DIM];
  __shared__ float zsh[256];
  __shared__ float zred[128];
  const int tid = threadIdx.x;
  const int rid = tid >> 1, kh = tid & 1;
  const int hbase = blockIdx.x * 32;
  const int gate = rid >> 5, j = rid & 31;
  const int row = gate * H_DIM + hbase + j;          // row in [0, 2048)
  const unsigned short* wr = Whh + (size_t)row * H_DIM + kh * 256;
  float creg = 0.f;
  for (int t = 0; t < T_FRAMES; ++t) {
    hloc[tid]       = hcur[tid];
    hloc[tid + 256] = hcur[tid + 256];
    __syncthreads();
    float acc = 0.f;
    const float* hp = hloc + kh * 256;
#pragma unroll 8
    for (int k = 0; k < 256; ++k) acc = fmaf(bf2f(wr[k]), hp[k], acc);
    zsh[tid] = acc;
    __syncthreads();
    if (kh == 0) zred[rid] = zsh[tid] + zsh[tid + 1] + gates[(size_t)t * G_ROWS + row];
    __syncthreads();
    if (tid < 32) {
      float iv = sigm(zred[tid]);
      float fv = sigm(zred[32 + tid]);
      float gv = tanhf(zred[64 + tid]);
      float ov = sigm(zred[96 + tid]);
      creg = fv * creg + iv * gv;
      float hv = ov * tanhf(creg);
      hcur[hbase + tid] = hv;
      hout[(size_t)t * H_DIM + hbase + tid] = hv;
    }
    __threadfence();
    global_barrier(cnt, gen, LSTM_BLOCKS);
  }
}

// ---------------- host launcher ----------------

extern "C" void kernel_launch(void* const* d_in, const int* in_sizes, int n_in,
                              void* d_out, int out_size, void* d_ws, size_t ws_size,
                              hipStream_t stream)
{
  (void)in_sizes; (void)n_in; (void)out_size; (void)ws_size;
  const float* x    = (const float*)d_in[0];
  const float* W1   = (const float*)d_in[1];
  const float* b1   = (const float*)d_in[2];
  const float* W2   = (const float*)d_in[3];
  const float* b2   = (const float*)d_in[4];
  const float* Wih  = (const float*)d_in[5];
  const float* Whh  = (const float*)d_in[6];
  const float* bih  = (const float*)d_in[7];
  const float* bhh  = (const float*)d_in[8];
  // d_in[9], d_in[10] = W3, b3: dead (softmax over size-1 dim == 1)
  const float* W4   = (const float*)d_in[11];
  const float* b4   = (const float*)d_in[12];
  const float* Wcls = (const float*)d_in[13];
  const float* bcls = (const float*)d_in[14];
  float* out = (float*)d_out;

  char* base = (char*)d_ws;
  size_t off = 0;
  auto alloc = [&](size_t bytes) -> char* {
    char* p = base + off;
    off = (off + bytes + 255) & ~(size_t)255;
    return p;
  };

  float*    scalars = (float*)alloc(64);                 // [0]=alpha_sum [1]=s2 [2]=s4
  unsigned* ctrl    = (unsigned*)alloc(64);              // barrier cnt/gen
  float*    hcur    = (float*)alloc(H_DIM * 4);
  float*    alpha   = (float*)alloc(T_FRAMES * 4);
  float*    beta    = (float*)alloc(T_FRAMES * 4);
  float*    gammav  = (float*)alloc(T_FRAMES * 4);
  float*    zacc    = (float*)alloc(D_FEAT * 4);
  float*    wsum    = (float*)alloc(D_FEAT * 4);
  float*    biassum = (float*)alloc(G_ROWS * 4);
  unsigned short* wihbf  = (unsigned short*)alloc((size_t)G_ROWS * D_FEAT * 2);
  unsigned short* whhbf  = (unsigned short*)alloc((size_t)G_ROWS * H_DIM * 2);
  unsigned short* wclsbf = (unsigned short*)alloc((size_t)C_CLS * H_DIM * 2);
  float*          wbuf   = (float*)alloc((size_t)T_FRAMES * D_FEAT * 4);
  unsigned short* wbf    = (unsigned short*)alloc((size_t)T_FRAMES * D_FEAT * 2);
  float*          gates  = (float*)alloc((size_t)T_FRAMES * G_ROWS * 4);
  float*          hout   = (float*)alloc((size_t)T_FRAMES * H_DIM * 4);
  unsigned short* cbf    = (unsigned short*)alloc((size_t)T_FRAMES * H_DIM * 2);

  k_init<<<1, 512, 0, stream>>>(scalars, ctrl, hcur);
  k_f32_to_bf16<<<(G_ROWS * D_FEAT) / 256, 256, 0, stream>>>(Wih, wihbf, G_ROWS * D_FEAT);
  k_f32_to_bf16<<<(G_ROWS * H_DIM) / 256, 256, 0, stream>>>(Whh, whhbf, G_ROWS * H_DIM);
  k_f32_to_bf16<<<(C_CLS * H_DIM) / 256, 256, 0, stream>>>(Wcls, wclsbf, C_CLS * H_DIM);
  k_biassum<<<G_ROWS / 256, 256, 0, stream>>>(bih, bhh, biassum, G_ROWS);

  // alpha, alpha_sum, z'-accum, s2, beta
  k_rowdot_sigmoid<<<T_FRAMES / 8, 256, 0, stream>>>(x, W1, nullptr, b1, alpha, D_FEAT);
  k_reduce_sum<<<1, 256, 0, stream>>>(alpha, T_FRAMES, &scalars[0]);
  k_zacc<<<D_FEAT / 256, 256, 0, stream>>>(x, alpha, zacc);
  k_dot_div_bias<<<1, 256, 0, stream>>>(zacc, W2 + D_FEAT, D_FEAT, &scalars[0], 1.f, b2, &scalars[1]);
  k_rowdot_sigmoid<<<T_FRAMES / 8, 256, 0, stream>>>(x, W2, &scalars[1], nullptr, beta, D_FEAT);

  // w scan, s4, gamma
  k_wscan<<<D_FEAT / 256, 256, 0, stream>>>(x, beta, wbuf, wbf, wsum);
  k_dot_div_bias<<<1, 256, 0, stream>>>(wsum, W4 + D_FEAT, D_FEAT, nullptr, (float)T_FRAMES, b4, &scalars[2]);
  k_rowdot_sigmoid<<<T_FRAMES / 8, 256, 0, stream>>>(wbuf, W4, &scalars[2], nullptr, gammav, D_FEAT);

  // LSTM input projection (big WMMA GEMM), recurrence, c scan, classifier GEMM
  dim3 g1(G_ROWS / 128, T_FRAMES / 64);
  k_gemm_bf16_tn<<<g1, 256, 0, stream>>>(wbf, wihbf, biassum, gates, T_FRAMES, G_ROWS, D_FEAT);
  k_lstm<<<LSTM_BLOCKS, 256, 0, stream>>>(gates, whhbf, hcur, hout, &ctrl[0], &ctrl[1]);
  k_cscan<<<H_DIM / 256, 256, 0, stream>>>(hout, gammav, cbf);
  dim3 g2((C_CLS + 127) / 128, T_FRAMES / 64);
  k_gemm_bf16_tn<<<g2, 256, 0, stream>>>(cbf, wclsbf, bcls, out, T_FRAMES, C_CLS, H_DIM);
}
